// MultiheadAttention_20023137534096
// MI455X (gfx1250) — matrix operand order repaired and checked by reading, not compiled, not run on hardware
//
#include <hip/hip_runtime.h>

// ---------------------------------------------------------------------------
// MI455X (gfx1250) multi-head attention: bf16 WMMA everywhere, fp32 accum.
//   B=4, S=2048, D=768, H=12, hd=64
// ---------------------------------------------------------------------------

typedef __attribute__((ext_vector_type(16))) __bf16 v16bf;
typedef __attribute__((ext_vector_type(8)))  __bf16 v8bf;
typedef __attribute__((ext_vector_type(8)))  float  v8f;

#define B_DIM   4
#define S_LEN   2048
#define D_MODEL 768
#define NH      12
#define HD      64
#define MROWS   (B_DIM * S_LEN)   // 8192
#define E3D     (3 * D_MODEL)     // 2304

__device__ __forceinline__ v8f wmma_bf16(v16bf a, v16bf b, v8f c) {
  // D = A(16x32) * B(32x16) + C, f32 accumulate
  return __builtin_amdgcn_wmma_f32_16x16x32_bf16(
      /*neg_a=*/false, a, /*neg_b=*/false, b,
      /*c_mod=*/(short)0, c, /*reuse_a=*/false, /*reuse_b=*/false);
}

// A-fragment: lane l holds row l%16; K = {16g..16g+15}, g = l/16.
// Built from two contiguous 16B chunks.
template <typename P>
__device__ __forceinline__ v16bf afrag(const P* p0, const P* p1) {
  v8bf lo = *(const v8bf*)p0;
  v8bf hi = *(const v8bf*)p1;
  v16bf a;
#pragma unroll
  for (int i = 0; i < 8; ++i) { a[i] = lo[i]; a[i + 8] = hi[i]; }
  return a;
}

// ---------------------------------------------------------------------------
// fp32 -> bf16 convert (grid-stride)
// ---------------------------------------------------------------------------
__global__ void mha_f32_to_bf16(const float* __restrict__ in,
                                __bf16* __restrict__ out, int n) {
  int i = blockIdx.x * blockDim.x + threadIdx.x;
  int stride = gridDim.x * blockDim.x;
  for (; i < n; i += stride) out[i] = (__bf16)in[i];
}

// ---------------------------------------------------------------------------
// QKV projection: qkv[m,e] = sum_d x[m,d] * w_qkv[e,d] + b_qkv[e]
// Each wave computes a 16(M) x 64(N) tile. Epilogue routes into:
//   Q: [b,h,s,f] bf16, pre-scaled by 1/sqrt(hd)=0.125
//   K: [b,h,s,f] bf16  (row-major serves QK^T B-fragments)
//   V: [b,h,f,s] bf16  (transposed serves PV B-fragments)
// ---------------------------------------------------------------------------
__global__ __launch_bounds__(256) void mha_gemm_qkv(
    const __bf16* __restrict__ x16, const __bf16* __restrict__ w16,
    const float* __restrict__ bqkv, __bf16* __restrict__ qb,
    __bf16* __restrict__ kbuf, __bf16* __restrict__ vtb) {
  const int lane  = threadIdx.x & 31;
  const int wave  = threadIdx.x >> 5;
  const int tile  = blockIdx.x * 8 + wave;
  const int NT    = E3D / 64;  // 36
  const int nt    = tile % NT;
  const int mt    = tile / NT;
  const int row16 = lane & 15;
  const int half  = lane >> 4;

  v8f acc[4] = {{}, {}, {}, {}};
  const __bf16* arow  = x16 + (size_t)(mt * 16 + row16) * D_MODEL;
  const __bf16* wbase = w16 + (size_t)(nt * 64 + row16) * D_MODEL + 16 * half;

#pragma unroll 4
  for (int k0 = 0; k0 < D_MODEL; k0 += 32) {
    v16bf af = afrag(arow + k0 + 16 * half, arow + k0 + 16 * half + 8);
#pragma unroll
    for (int j = 0; j < 4; ++j)
      acc[j] = wmma_bf16(af, *(const v16bf*)(wbase + (size_t)j * 16 * D_MODEL + k0),
                         acc[j]);
  }

#pragma unroll
  for (int j = 0; j < 4; ++j) {
    const int e = nt * 64 + j * 16 + row16;
    const float be = bqkv[e];
#pragma unroll
    for (int r = 0; r < 8; ++r) {
      const int m = mt * 16 + r + 8 * half;
      const float v = acc[j][r] + be;
      const int b = m >> 11;       // m / 2048
      const int s = m & 2047;
      if (e < D_MODEL) {
        const int h = e >> 6, f = e & 63;
        qb[(((size_t)b * NH + h) * S_LEN + s) * HD + f] = (__bf16)(v * 0.125f);
      } else if (e < 2 * D_MODEL) {
        const int e2 = e - D_MODEL, h = e2 >> 6, f = e2 & 63;
        kbuf[(((size_t)b * NH + h) * S_LEN + s) * HD + f] = (__bf16)v;
      } else {
        const int e2 = e - 2 * D_MODEL, h = e2 >> 6, f = e2 & 63;
        vtb[(((size_t)b * NH + h) * HD + f) * S_LEN + s] = (__bf16)v;
      }
    }
  }
}

// ---------------------------------------------------------------------------
// Flash attention: each wave owns 16 query rows of one (b,h); loops keys in
// blocks of 32. Scores: 4 WMMAs (hd=64 -> 2 K-chunks x 2 key subtiles).
// Online softmax via 16-lane shuffle reductions (C-layout rows r+8*half).
// P tile bounced through 1KB LDS per wave to re-layout C->A fragment, then
// 4 WMMAs against transposed V. Output written bf16 [b,s,D].
// ---------------------------------------------------------------------------
__global__ __launch_bounds__(256) void mha_flash_attn(
    const __bf16* __restrict__ qb, const __bf16* __restrict__ kbuf,
    const __bf16* __restrict__ vtb, __bf16* __restrict__ valb) {
  __shared__ __align__(16) __bf16 lds_p[8 * 16 * 32];
  const int lane  = threadIdx.x & 31;
  const int wave  = threadIdx.x >> 5;
  const int gwave = blockIdx.x * 8 + wave;
  const int mt    = gwave & (S_LEN / 16 - 1);  // 128 q-tiles per (b,h)
  const int bh    = gwave >> 7;
  const int row16 = lane & 15;
  const int half  = lane >> 4;
  const int m0    = mt * 16;

  const __bf16* q  = qb   + (size_t)bh * S_LEN * HD;
  const __bf16* k  = kbuf + (size_t)bh * S_LEN * HD;
  const __bf16* vt = vtb  + (size_t)bh * HD * S_LEN;

  const __bf16* qrow = q + (size_t)(m0 + row16) * HD;
  const v16bf qa0 = afrag(qrow + 16 * half,      qrow + 8 + 16 * half);
  const v16bf qa1 = afrag(qrow + 32 + 16 * half, qrow + 40 + 16 * half);

  v8f oacc[4] = {{}, {}, {}, {}};
  float mrun[8], lrun[8];
#pragma unroll
  for (int r = 0; r < 8; ++r) { mrun[r] = -3.0e38f; lrun[r] = 0.0f; }

  __bf16* lp = lds_p + wave * (16 * 32);

  for (int kb0 = 0; kb0 < S_LEN; kb0 += 32) {
    // ---- scores: two 16-key subtiles, K-dim = hd = 64 (2 chunks) ----
    const __bf16* kr0 = k + (size_t)(kb0 + row16) * HD + 16 * half;
    const __bf16* kr1 = k + (size_t)(kb0 + 16 + row16) * HD + 16 * half;
    v8f s0 = {};
    s0 = wmma_bf16(qa0, *(const v16bf*)(kr0), s0);
    s0 = wmma_bf16(qa1, *(const v16bf*)(kr0 + 32), s0);
    v8f s1 = {};
    s1 = wmma_bf16(qa0, *(const v16bf*)(kr1), s1);
    s1 = wmma_bf16(qa1, *(const v16bf*)(kr1 + 32), s1);

    // ---- online softmax: row max / exp / row sum (rows r+8*half) ----
    float smax[8];
#pragma unroll
    for (int r = 0; r < 8; ++r) smax[r] = fmaxf(s0[r], s1[r]);
#pragma unroll
    for (int d = 1; d < 16; d <<= 1) {
#pragma unroll
      for (int r = 0; r < 8; ++r)
        smax[r] = fmaxf(smax[r], __shfl_xor(smax[r], d, 32));
    }
    float ps0[8], ps1[8], rsum[8], scale[8];
#pragma unroll
    for (int r = 0; r < 8; ++r) {
      const float mn = fmaxf(mrun[r], smax[r]);
      scale[r] = __expf(mrun[r] - mn);
      ps0[r]   = __expf(s0[r] - mn);
      ps1[r]   = __expf(s1[r] - mn);
      rsum[r]  = ps0[r] + ps1[r];
      mrun[r]  = mn;
    }
#pragma unroll
    for (int d = 1; d < 16; d <<= 1) {
#pragma unroll
      for (int r = 0; r < 8; ++r) rsum[r] += __shfl_xor(rsum[r], d, 32);
    }
#pragma unroll
    for (int r = 0; r < 8; ++r) {
      lrun[r] = lrun[r] * scale[r] + rsum[r];
      oacc[0][r] *= scale[r];
      oacc[1][r] *= scale[r];
      oacc[2][r] *= scale[r];
      oacc[3][r] *= scale[r];
      const int row = r + 8 * half;
      lp[row * 32 + row16]      = (__bf16)ps0[r];
      lp[row * 32 + 16 + row16] = (__bf16)ps1[r];
    }
    // same-wave LDS producer->consumer: only the DS counter wait is needed
    asm volatile("s_wait_dscnt 0" ::: "memory");

    // ---- P (16x32) as A-fragment from LDS, PV over 4 feature tiles ----
    const __bf16* prow = lp + row16 * 32;
    const v16bf pa = afrag(prow + 16 * half, prow + 8 + 16 * half);
#pragma unroll
    for (int t = 0; t < 4; ++t) {
      const __bf16* vp = vt + (size_t)(t * 16 + row16) * S_LEN + kb0 + 16 * half;
      oacc[t] = wmma_bf16(pa, *(const v16bf*)vp, oacc[t]);
    }
  }

  // ---- normalize and write [b, s, h*64+f] bf16 ----
  const int b = bh / NH, h = bh % NH;
#pragma unroll
  for (int r = 0; r < 8; ++r) {
    const int row = r + 8 * half;
    const float inv = 1.0f / lrun[r];
    const size_t obase = ((size_t)b * S_LEN + m0 + row) * D_MODEL + h * HD;
#pragma unroll
    for (int t = 0; t < 4; ++t)
      valb[obase + t * 16 + row16] = (__bf16)(oacc[t][r] * inv);
  }
}

// ---------------------------------------------------------------------------
// Output projection: out[m,n] = sum_d val[m,d] * w_out[n,d] + b_out[n], fp32 out
// ---------------------------------------------------------------------------
__global__ __launch_bounds__(256) void mha_gemm_out(
    const __bf16* __restrict__ v16x, const __bf16* __restrict__ w16,
    const float* __restrict__ bout, float* __restrict__ out) {
  const int lane  = threadIdx.x & 31;
  const int wave  = threadIdx.x >> 5;
  const int tile  = blockIdx.x * 8 + wave;
  const int NT    = D_MODEL / 64;  // 12
  const int nt    = tile % NT;
  const int mt    = tile / NT;
  const int row16 = lane & 15;
  const int half  = lane >> 4;

  v8f acc[4] = {{}, {}, {}, {}};
  const __bf16* arow  = v16x + (size_t)(mt * 16 + row16) * D_MODEL;
  const __bf16* wbase = w16 + (size_t)(nt * 64 + row16) * D_MODEL + 16 * half;

#pragma unroll 4
  for (int k0 = 0; k0 < D_MODEL; k0 += 32) {
    v16bf af = afrag(arow + k0 + 16 * half, arow + k0 + 16 * half + 8);
#pragma unroll
    for (int j = 0; j < 4; ++j)
      acc[j] = wmma_bf16(af, *(const v16bf*)(wbase + (size_t)j * 16 * D_MODEL + k0),
                         acc[j]);
  }

#pragma unroll
  for (int j = 0; j < 4; ++j) {
    const int e = nt * 64 + j * 16 + row16;
    const float be = bout[e];
#pragma unroll
    for (int r = 0; r < 8; ++r) {
      const int m = mt * 16 + r + 8 * half;
      out[(size_t)m * D_MODEL + e] = acc[j][r] + be;
    }
  }
}

// ---------------------------------------------------------------------------
extern "C" void kernel_launch(void* const* d_in, const int* in_sizes, int n_in,
                              void* d_out, int out_size, void* d_ws, size_t ws_size,
                              hipStream_t stream) {
  (void)in_sizes; (void)n_in; (void)out_size; (void)ws_size;
  const float* x     = (const float*)d_in[0];
  const float* w_qkv = (const float*)d_in[1];
  const float* b_qkv = (const float*)d_in[2];
  const float* w_out = (const float*)d_in[3];
  const float* b_out = (const float*)d_in[4];
  float* out = (float*)d_out;

  char* ws = (char*)d_ws;
  size_t off = 0;
  auto take = [&](size_t bytes) {
    char* p = ws + off;
    off += (bytes + 255) & ~(size_t)255;
    return p;
  };
  __bf16* x16    = (__bf16*)take((size_t)MROWS * D_MODEL * 2);   // 12.6 MB
  __bf16* wqkv16 = (__bf16*)take((size_t)E3D * D_MODEL * 2);     //  3.5 MB
  __bf16* wout16 = (__bf16*)take((size_t)D_MODEL * D_MODEL * 2); //  1.2 MB
  __bf16* qb     = (__bf16*)take((size_t)MROWS * D_MODEL * 2);   // 12.6 MB
  __bf16* kb     = (__bf16*)take((size_t)MROWS * D_MODEL * 2);   // 12.6 MB
  __bf16* vtb    = (__bf16*)take((size_t)MROWS * D_MODEL * 2);   // 12.6 MB
  __bf16* valb   = (__bf16*)take((size_t)MROWS * D_MODEL * 2);   // 12.6 MB

  {
    int n = MROWS * D_MODEL;
    mha_f32_to_bf16<<<(n + 255) / 256, 256, 0, stream>>>(x, x16, n);
  }
  {
    int n = E3D * D_MODEL;
    mha_f32_to_bf16<<<(n + 255) / 256, 256, 0, stream>>>(w_qkv, wqkv16, n);
  }
  {
    int n = D_MODEL * D_MODEL;
    mha_f32_to_bf16<<<(n + 255) / 256, 256, 0, stream>>>(w_out, wout16, n);
  }

  // (8192/16)*(2304/64) = 18432 wave-tiles / 8 waves = 2304 blocks
  mha_gemm_qkv<<<(MROWS / 16) * (E3D / 64) / 8, 256, 0, stream>>>(
      x16, wqkv16, b_qkv, qb, kb, vtb);

  // 4*12*128 = 6144 waves / 8 = 768 blocks
  mha_flash_attn<<<(B_DIM * NH * (S_LEN / 16)) / 8, 256, 0, stream>>>(
      qb, kb, vtb, valb);

  // (8192/16)*(768/64) = 6144 wave-tiles / 8 = 768 blocks
  mha_gemm_out<<<(MROWS / 16) * (D_MODEL / 64) / 8, 256, 0, stream>>>(
      valb, wout16, b_out, out);
}
